// Router_16965120819864
// MI455X (gfx1250) — matrix operand order repaired and checked by reading, not compiled, not run on hardware
//
#include <hip/hip_runtime.h>
#include <hip/hip_bf16.h>
#include <math.h>
#include <stdint.h>

// ---------------- constants (shapes fixed by the reference) ----------------
#define C_DIM 2048      // embedding dim
#define N_EXP 64        // experts
#define TOPK 2
#define TOK_PER_BLOCK 128
#define THREADS 128     // 4 waves; each wave owns 32 tokens (2 M-tiles)
#define KC 128          // K-chunk staged into LDS for B

typedef __attribute__((ext_vector_type(16))) __bf16 v16bf;
typedef __attribute__((ext_vector_type(8)))  float  v8f;
typedef __attribute__((ext_vector_type(4)))  float  v4f;

// ---------------- kernel W: w_g fp32 -> bf16 ----------------
__global__ void convert_wg(const float* __restrict__ wg,
                           __bf16* __restrict__ wgb, int total) {
    int i = blockIdx.x * blockDim.x + threadIdx.x;
    if (i < total) wgb[i] = (__bf16)wg[i];
}

// Build an A fragment in contiguous K order (matches B):
// lanes 0-15: K 0-15 ; lanes 16-31: K 16-31
__device__ __forceinline__ v16bf make_a_frag(const float* xrow, int k0, int khalf) {
    v4f a0 = *(const v4f*)(xrow + k0 + 2 * khalf);
    v4f a1 = *(const v4f*)(xrow + k0 + 2 * khalf + 4);
    v4f a2 = *(const v4f*)(xrow + k0 + 2 * khalf + 8);
    v4f a3 = *(const v4f*)(xrow + k0 + 2 * khalf + 12);
    v16bf a;
#pragma unroll
    for (int i = 0; i < 4; ++i) {
        a[i]      = (__bf16)a0[i];
        a[4 + i]  = (__bf16)a1[i];
        a[8 + i]  = (__bf16)a2[i];
        a[12 + i] = (__bf16)a3[i];
    }
    return a;
}

// ---------------- kernel A: GEMM (WMMA bf16) + top-2 + local ranks ----------------
__global__ __launch_bounds__(THREADS) void router_gemm_topk(
    const float* __restrict__ x, const __bf16* __restrict__ wgb,
    int* __restrict__ top_idx, float* __restrict__ top_prob,
    int* __restrict__ local_rank, int* __restrict__ cnt0,
    int* __restrict__ cnt1) {
    __shared__ __align__(64) __bf16 sB[N_EXP * KC];           // 16 KB
    __shared__ float sLogits[TOK_PER_BLOCK * N_EXP];          // 32 KB
    __shared__ int   sIdx[TOK_PER_BLOCK * TOPK];
    __shared__ int   sCnt[TOPK * N_EXP];
    __shared__ int   sLr[TOK_PER_BLOCK * TOPK];

    const int tid   = threadIdx.x;
    const int wave  = tid >> 5;            // 0..3
    const int lane  = tid & 31;
    const int trow  = lane & 15;           // M (A) or N (B) index within tile
    const int hi    = lane >> 4;           // lane half
    const int khalf = hi ? 8 : 0;
    const int tblock = blockIdx.x * TOK_PER_BLOCK;

    v8f acc[2][4];
#pragma unroll
    for (int mt = 0; mt < 2; ++mt)
#pragma unroll
        for (int nt = 0; nt < 4; ++nt) acc[mt][nt] = v8f{};

    const float* xrow0 = x + (size_t)(tblock + wave * 32 + trow) * C_DIM;
    const float* xrow1 = xrow0 + (size_t)16 * C_DIM;

    for (int kc = 0; kc < C_DIM; kc += KC) {
        __syncthreads();  // all waves done reading previous sB
        // async-stage B chunk [N_EXP x KC] bf16 -> LDS: 1024 x 16B units
        for (int u = tid; u < (N_EXP * KC) / 8; u += THREADS) {
            unsigned lds32 =
                (unsigned)(uintptr_t)(&sB[u * 8]);  // flat low 32 bits == LDS addr
            unsigned long long gaddr = (unsigned long long)(uintptr_t)(
                wgb + (size_t)(u >> 4) * C_DIM + kc + (u & 15) * 8);
            asm volatile("global_load_async_to_lds_b128 %0, %1, off"
                         :
                         : "v"(lds32), "v"(gaddr)
                         : "memory");
        }
        asm volatile("s_wait_asynccnt 0x0" ::: "memory");
        if (kc + KC < C_DIM) __builtin_prefetch(xrow0 + kc + KC, 0, 0);
        __syncthreads();

        for (int ks = 0; ks < KC; ks += 32) {
            const int k0 = kc + ks;
            v16bf a0 = make_a_frag(xrow0, k0, khalf);
            v16bf a1 = make_a_frag(xrow1, k0, khalf);
            // B fragments: lane = N, low lane-half K 0-15, high half K 16-31
            const __bf16* bbase = &sB[trow * KC + ks + (hi ? 16 : 0)];
            v16bf b0 = *(const v16bf*)(bbase + 0 * 16 * KC);
            v16bf b1 = *(const v16bf*)(bbase + 1 * 16 * KC);
            v16bf b2 = *(const v16bf*)(bbase + 2 * 16 * KC);
            v16bf b3 = *(const v16bf*)(bbase + 3 * 16 * KC);
            acc[0][0] = __builtin_amdgcn_wmma_f32_16x16x32_bf16(
                false, a0, false, b0, (short)0, acc[0][0], false, false);
            acc[1][0] = __builtin_amdgcn_wmma_f32_16x16x32_bf16(
                false, a1, false, b0, (short)0, acc[1][0], false, false);
            acc[0][1] = __builtin_amdgcn_wmma_f32_16x16x32_bf16(
                false, a0, false, b1, (short)0, acc[0][1], false, false);
            acc[1][1] = __builtin_amdgcn_wmma_f32_16x16x32_bf16(
                false, a1, false, b1, (short)0, acc[1][1], false, false);
            acc[0][2] = __builtin_amdgcn_wmma_f32_16x16x32_bf16(
                false, a0, false, b2, (short)0, acc[0][2], false, false);
            acc[1][2] = __builtin_amdgcn_wmma_f32_16x16x32_bf16(
                false, a1, false, b2, (short)0, acc[1][2], false, false);
            acc[0][3] = __builtin_amdgcn_wmma_f32_16x16x32_bf16(
                false, a0, false, b3, (short)0, acc[0][3], false, false);
            acc[1][3] = __builtin_amdgcn_wmma_f32_16x16x32_bf16(
                false, a1, false, b3, (short)0, acc[1][3], false, false);
        }
    }
    __syncthreads();

    // spill logits: VGPR r, lanes 0-15 -> M=r ; lanes 16-31 -> M=8+r ; N=trow
#pragma unroll
    for (int mt = 0; mt < 2; ++mt)
#pragma unroll
        for (int nt = 0; nt < 4; ++nt)
#pragma unroll
            for (int r = 0; r < 8; ++r)
                sLogits[(wave * 32 + mt * 16 + r + khalf) * N_EXP + nt * 16 + trow] =
                    acc[mt][nt][r];
    __syncthreads();

    // top-2 + softmax per token (one token per thread)
    if (tid < TOK_PER_BLOCK) {
        const float* lg = &sLogits[tid * N_EXP];
        float b0 = -3.4e38f, b1 = -3.4e38f;
        int i0 = 0, i1 = 0;
        for (int e = 0; e < N_EXP; ++e) {
            float v = lg[e];
            if (v > b0) { b1 = b0; i1 = i0; b0 = v; i0 = e; }
            else if (v > b1) { b1 = v; i1 = e; }
        }
        float p0 = 1.0f / (1.0f + expf(b1 - b0));
        float p1 = 1.0f - p0;
        int n = tblock + tid;
        sIdx[tid * 2 + 0] = i0;
        sIdx[tid * 2 + 1] = i1;
        top_idx[n * 2 + 0] = i0;
        top_idx[n * 2 + 1] = i1;
        top_prob[n * 2 + 0] = p0;
        top_prob[n * 2 + 1] = p1;
    }
    if (tid < TOPK * N_EXP) sCnt[tid] = 0;
    __syncthreads();

    // deterministic token-order local ranks within block (serial, tiny)
    if (tid == 0) {
        for (int nn = 0; nn < TOK_PER_BLOCK; ++nn)
            for (int k = 0; k < TOPK; ++k) {
                int e = sIdx[nn * 2 + k];
                sLr[nn * 2 + k] = sCnt[k * N_EXP + e]++;
            }
    }
    __syncthreads();

    if (tid < TOK_PER_BLOCK) {
        int n = tblock + tid;
        local_rank[n * 2 + 0] = sLr[tid * 2 + 0];
        local_rank[n * 2 + 1] = sLr[tid * 2 + 1];
    }
    if (tid < N_EXP) {
        cnt0[blockIdx.x * N_EXP + tid] = sCnt[tid];
        cnt1[blockIdx.x * N_EXP + tid] = sCnt[N_EXP + tid];
    }
}

// ---------------- kernel B: exclusive scan over block counts ----------------
__global__ void router_scan(const int* __restrict__ cnt0,
                            const int* __restrict__ cnt1,
                            int* __restrict__ pre0, int* __restrict__ pre1,
                            int* __restrict__ tot0, int nblocks) {
    int e = threadIdx.x;
    if (e >= N_EXP) return;
    int run = 0;
    for (int b = 0; b < nblocks; ++b) { pre0[b * N_EXP + e] = run; run += cnt0[b * N_EXP + e]; }
    tot0[e] = run;
    run = 0;
    for (int b = 0; b < nblocks; ++b) { pre1[b * N_EXP + e] = run; run += cnt1[b * N_EXP + e]; }
}

// ---------------- kernel C1: ranks / capacity / small outputs ----------------
__global__ void router_finalize(const int* __restrict__ top_idx,
                                const float* __restrict__ top_prob,
                                const int* __restrict__ local_rank,
                                const int* __restrict__ pre0,
                                const int* __restrict__ pre1,
                                const int* __restrict__ tot0,
                                int* __restrict__ keep,
                                float* __restrict__ out_probs,
                                float* __restrict__ out_idx,
                                float* __restrict__ out_rank,
                                int num_tokens, int capacity) {
    int i = blockIdx.x * blockDim.x + threadIdx.x;
    if (i >= num_tokens * TOPK) return;
    int n = i >> 1, k = i & 1;
    int b = n / TOK_PER_BLOCK;
    int e = top_idx[i];
    int pos = local_rank[i] +
              (k ? (tot0[e] + pre1[b * N_EXP + e]) : pre0[b * N_EXP + e]);
    int kp = pos < capacity;
    keep[i] = kp;
    out_probs[i] = kp ? top_prob[i] : 0.0f;
    out_idx[i]   = (float)e;
    out_rank[i]  = (float)pos;   // (position-1)*onehot summed == pos (0-based)
}

// ---------------- kernel C2: scatter one-hot capacity-gated mask ----------------
__global__ void router_write_mask(const int* __restrict__ top_idx,
                                  const int* __restrict__ keep,
                                  float* __restrict__ out_mask, int total) {
    int i = blockIdx.x * blockDim.x + threadIdx.x;
    if (i >= total) return;
    int e  = i & (N_EXP - 1);
    int nk = i >> 6;  // n*2 + k  (mask layout [N, k, E] row-major)
    out_mask[i] = (top_idx[nk] == e && keep[nk]) ? 1.0f : 0.0f;
}

// ---------------- host launcher ----------------
extern "C" void kernel_launch(void* const* d_in, const int* in_sizes, int n_in,
                              void* d_out, int out_size, void* d_ws, size_t ws_size,
                              hipStream_t stream) {
    const float* x  = (const float*)d_in[0];
    const float* wg = (const float*)d_in[1];

    const int C = in_sizes[1] / N_EXP;          // 2048
    const int N = in_sizes[0] / C;              // 32768 tokens
    const int nblocks = N / TOK_PER_BLOCK;      // 256

    int capacity = (int)floor((double)TOPK * 1.25 * (double)N / (double)N_EXP);
    capacity += capacity % 2;
    if (capacity < 8) capacity = 8;

    // workspace carve-up (256B aligned)
    char* p = (char*)d_ws;
    auto carve = [&](size_t bytes) { char* r = p; p += (bytes + 255) & ~(size_t)255; return r; };
    __bf16* wgb       = (__bf16*)carve((size_t)N_EXP * C * sizeof(__bf16));
    int*    top_idx   = (int*)   carve((size_t)N * TOPK * sizeof(int));
    float*  top_prob  = (float*) carve((size_t)N * TOPK * sizeof(float));
    int*    local_rk  = (int*)   carve((size_t)N * TOPK * sizeof(int));
    int*    cnt0      = (int*)   carve((size_t)nblocks * N_EXP * sizeof(int));
    int*    cnt1      = (int*)   carve((size_t)nblocks * N_EXP * sizeof(int));
    int*    pre0      = (int*)   carve((size_t)nblocks * N_EXP * sizeof(int));
    int*    pre1      = (int*)   carve((size_t)nblocks * N_EXP * sizeof(int));
    int*    tot0      = (int*)   carve((size_t)N_EXP * sizeof(int));
    int*    keep      = (int*)   carve((size_t)N * TOPK * sizeof(int));

    // output layout (floats, reference return order)
    float* out_mask  = (float*)d_out;                        // N*2*64
    float* out_probs = out_mask + (size_t)N * TOPK * N_EXP;  // N*2
    float* out_idx   = out_probs + (size_t)N * TOPK;         // N*2
    float* out_rank  = out_idx + (size_t)N * TOPK;           // N*2

    int wtotal = N_EXP * C;
    convert_wg<<<(wtotal + 255) / 256, 256, 0, stream>>>(wg, wgb, wtotal);

    router_gemm_topk<<<nblocks, THREADS, 0, stream>>>(
        x, wgb, top_idx, top_prob, local_rk, cnt0, cnt1);

    router_scan<<<1, N_EXP, 0, stream>>>(cnt0, cnt1, pre0, pre1, tot0, nblocks);

    int small = N * TOPK;
    router_finalize<<<(small + 255) / 256, 256, 0, stream>>>(
        top_idx, top_prob, local_rk, pre0, pre1, tot0,
        keep, out_probs, out_idx, out_rank, N, capacity);

    int mtotal = N * TOPK * N_EXP;
    router_write_mask<<<(mtotal + 255) / 256, 256, 0, stream>>>(
        top_idx, keep, out_mask, mtotal);
}
